// GQA_55027120996455
// MI455X (gfx1250) — matrix operand order repaired and checked by reading, not compiled, not run on hardware
//
#include <hip/hip_runtime.h>

#define HIDDEN 2048
#define HEADS 16
#define GROUPS 4
#define HEAD_DIM 128
#define KV_DIM 512
#define BATCH 2
#define SEQ 2048
#define MROWS (BATCH * SEQ)   // 4096

typedef __attribute__((ext_vector_type(16))) __bf16 v16bf;
typedef __attribute__((ext_vector_type(8)))  float  v8f;

union Frag16 { uint4 u4[2]; v16bf v; };

__device__ __forceinline__ unsigned short f32_to_bf16(float f) {
  unsigned int u = __float_as_uint(f);
  u += 0x7FFFu + ((u >> 16) & 1u);   // round-to-nearest-even
  return (unsigned short)(u >> 16);
}

__device__ __forceinline__ v8f wmma_bf16(const Frag16& a, const Frag16& b, v8f c) {
  return __builtin_amdgcn_wmma_f32_16x16x32_bf16(false, a.v, false, b.v, (short)0, c,
                                                 false, false);
}

// LDS byte offset from a generic pointer to __shared__ (flat addr[31:0] == LDS offset)
__device__ __forceinline__ unsigned lds_off(const void* p) {
  return (unsigned)(uintptr_t)p;
}

// CDNA5 async memory->LDS copy, 16 bytes per lane, tracked by ASYNCcnt
__device__ __forceinline__ void async_b128(unsigned lds, const unsigned short* gsrc) {
  asm volatile("global_load_async_to_lds_b128 %0, %1, off"
               :: "v"(lds), "v"((unsigned long long)(uintptr_t)gsrc)
               : "memory");
}
__device__ __forceinline__ void wait_async0() {
  asm volatile("s_wait_asynccnt 0x0" ::: "memory");
}

// ---------------------------------------------------------------------------
// fp32 -> bf16 (straight, 4-wide)
// ---------------------------------------------------------------------------
__global__ void cvt_bf16_kernel(const float* __restrict__ in,
                                unsigned short* __restrict__ out,
                                long long n4) {
  long long i = (long long)blockIdx.x * blockDim.x + threadIdx.x;
  long long stride = (long long)gridDim.x * blockDim.x;
  for (; i < n4; i += stride) {
    float4 f = ((const float4*)in)[i];
    unsigned short us[4];
    us[0] = f32_to_bf16(f.x);
    us[1] = f32_to_bf16(f.y);
    us[2] = f32_to_bf16(f.z);
    us[3] = f32_to_bf16(f.w);
    ((uint2*)out)[i] = *(const uint2*)us;
  }
}

// ---------------------------------------------------------------------------
// fp32 [K,N] -> bf16 [N,K] (tiled transpose, 32x32 tiles, 256 threads)
// ---------------------------------------------------------------------------
__global__ __launch_bounds__(256)
void cvt_t_bf16_kernel(const float* __restrict__ in,
                       unsigned short* __restrict__ out,
                       int K, int N) {
  __shared__ __attribute__((aligned(16))) unsigned short t[32][33];
  const int tx = threadIdx.x & 31;
  const int ty = threadIdx.x >> 5;          // 0..7
  const int k0 = blockIdx.y * 32;
  const int n0 = blockIdx.x * 32;
#pragma unroll
  for (int j = 0; j < 4; ++j) {
    int kk = ty + j * 8;
    t[tx][kk] = f32_to_bf16(in[(size_t)(k0 + kk) * N + n0 + tx]);
  }
  __syncthreads();
#pragma unroll
  for (int j = 0; j < 4; ++j) {
    int nn = ty + j * 8;
    out[(size_t)(n0 + nn) * K + k0 + tx] = t[nn][tx];
  }
}

// ---------------------------------------------------------------------------
// bf16 [B*S, KV_DIM] -> bf16 Vt[(b*G+g)*128 + d][S]  (tiled transpose)
// grid: (S/32, HEAD_DIM/32, B*G)
// ---------------------------------------------------------------------------
__global__ __launch_bounds__(256)
void vtrans_kernel(const unsigned short* __restrict__ in,
                   unsigned short* __restrict__ out) {
  __shared__ __attribute__((aligned(16))) unsigned short t[32][33];
  const int tx = threadIdx.x & 31;
  const int ty = threadIdx.x >> 5;
  const int s0 = blockIdx.x * 32;
  const int d0 = blockIdx.y * 32;
  const int bg = blockIdx.z;                 // b*GROUPS + g
  const int b  = bg >> 2;
  const int g  = bg & 3;
#pragma unroll
  for (int j = 0; j < 4; ++j) {
    int ss = ty + j * 8;
    t[tx][ss] = in[(size_t)(b * SEQ + s0 + ss) * KV_DIM + g * HEAD_DIM + d0 + tx];
  }
  __syncthreads();
#pragma unroll
  for (int j = 0; j < 4; ++j) {
    int dd = ty + j * 8;
    out[((size_t)bg * HEAD_DIM + d0 + dd) * SEQ + s0 + tx] = t[dd][tx];
  }
}

// ---------------------------------------------------------------------------
// bf16 GEMM, f32 accumulate:  C[M,N] = A[M,K] @ Bt[N,K]^T + bias[N]
// BM=BN=128, BK=32, 256 thr = 8 waves (4x2), wave tile 32x64.
// Double-buffered LDS filled with async B128 copies; fragment loads hoisted
// ahead of an 8-WMMA burst per k-step.
// ---------------------------------------------------------------------------
#define BM 128
#define BN 128
#define BK 32

__global__ __launch_bounds__(256)
void gemm_bf16_kernel(const unsigned short* __restrict__ A,   // [M,K]
                      const unsigned short* __restrict__ Bt,  // [N,K]
                      const float* __restrict__ bias,
                      unsigned short* __restrict__ Cb,
                      float* __restrict__ Cf,
                      int M, int N, int K) {
  __shared__ __attribute__((aligned(16))) unsigned short sA[2][BM * BK];
  __shared__ __attribute__((aligned(16))) unsigned short sB[2][BN * BK];

  const int tid  = threadIdx.x;
  const int lane = tid & 31;
  const int wave = tid >> 5;
  const int wm   = wave >> 1;
  const int wn   = wave & 1;
  const int hl   = lane >> 4;
  const int l16  = lane & 15;

  const int rowBase = blockIdx.y * BM;
  const int colBase = blockIdx.x * BN;

  // staging geometry: each thread owns 16 consecutive elements of one tile row
  const int srow = tid >> 1;            // 0..127
  const int sseg = (tid & 1) * 16;      // 0 or 16 elements
  const unsigned short* gA = A  + (size_t)(rowBase + srow) * K + sseg;
  const unsigned short* gB = Bt + (size_t)(colBase + srow) * K + sseg;
  const unsigned sOff = (unsigned)(srow * BK + sseg) * 2u;  // byte offset in tile

  v8f acc[2][4] = {};

  // prologue: stage k0 = 0 into buffer 0
  {
    unsigned la = lds_off(&sA[0][0]) + sOff;
    unsigned lb = lds_off(&sB[0][0]) + sOff;
    async_b128(la,      gA);
    async_b128(la + 16, gA + 8);
    async_b128(lb,      gB);
    async_b128(lb + 16, gB + 8);
  }

  int cur = 0;
  for (int k0 = 0; k0 < K; k0 += BK) {
    wait_async0();
    __syncthreads();

    if (k0 + BK < K) {
      unsigned la = lds_off(&sA[cur ^ 1][0]) + sOff;
      unsigned lb = lds_off(&sB[cur ^ 1][0]) + sOff;
      async_b128(la,      gA + k0 + BK);
      async_b128(la + 16, gA + k0 + BK + 8);
      async_b128(lb,      gB + k0 + BK);
      async_b128(lb + 16, gB + k0 + BK + 8);
      if (k0 + 2 * BK < K) {
        __builtin_prefetch(gA + k0 + 2 * BK, 0, 1);
        __builtin_prefetch(gB + k0 + 2 * BK, 0, 1);
      }
    }

    const unsigned short* cA = sA[cur];
    const unsigned short* cB = sB[cur];

    // hoist ALL fragment loads, then an 8-WMMA burst (independent accumulators)
    Frag16 a[2], bfr[4];
#pragma unroll
    for (int mt = 0; mt < 2; ++mt) {
      int row = wm * 32 + mt * 16 + l16;
      a[mt].u4[0] = *(const uint4*)(cA + row * BK + hl * 8);
      a[mt].u4[1] = *(const uint4*)(cA + row * BK + 16 + hl * 8);
    }
#pragma unroll
    for (int nt = 0; nt < 4; ++nt) {
      int ncol = wn * 64 + nt * 16 + l16;
      bfr[nt].u4[0] = *(const uint4*)(cB + ncol * BK + hl * 8);
      bfr[nt].u4[1] = *(const uint4*)(cB + ncol * BK + 16 + hl * 8);
    }
#pragma unroll
    for (int nt = 0; nt < 4; ++nt)
#pragma unroll
      for (int mt = 0; mt < 2; ++mt)
        acc[mt][nt] = wmma_bf16(a[mt], bfr[nt], acc[mt][nt]);

    cur ^= 1;
  }

  if (Cf) {
#pragma unroll
    for (int mt = 0; mt < 2; ++mt)
#pragma unroll
      for (int nt = 0; nt < 4; ++nt)
#pragma unroll
        for (int i = 0; i < 8; ++i) {
          int row = rowBase + wm * 32 + mt * 16 + hl * 8 + i;
          int col = colBase + wn * 64 + nt * 16 + l16;
          Cf[(size_t)row * N + col] = acc[mt][nt][i] + bias[col];
        }
  } else {
#pragma unroll
    for (int mt = 0; mt < 2; ++mt)
#pragma unroll
      for (int nt = 0; nt < 4; ++nt)
#pragma unroll
        for (int i = 0; i < 8; ++i) {
          int row = rowBase + wm * 32 + mt * 16 + hl * 8 + i;
          int col = colBase + wn * 64 + nt * 16 + l16;
          Cb[(size_t)row * N + col] = f32_to_bf16(acc[mt][nt][i] + bias[col]);
        }
  }
}

// ---------------------------------------------------------------------------
// Flash attention (GQA), double-buffered async K/V tiles.
// blockIdx = (qtile, head, batch); 4 waves, each wave owns 16 q-rows.
// ---------------------------------------------------------------------------
#define AQT 64
#define AKT 64

__global__ __launch_bounds__(128)
void attn_kernel(const unsigned short* __restrict__ Qb,  // [4096, 2048]
                 const unsigned short* __restrict__ Kb,  // [4096, 512]
                 const unsigned short* __restrict__ Vt,  // [(b*G+g)*128 + d][S]
                 const int* __restrict__ mask,           // [B, S]
                 unsigned short* __restrict__ Ob) {      // [4096, 2048]
  __shared__ __attribute__((aligned(16))) unsigned short sK[2][AKT * HEAD_DIM];
  __shared__ __attribute__((aligned(16))) unsigned short sV[2][HEAD_DIM * AKT];
  __shared__ __attribute__((aligned(16))) unsigned short sP[4][16 * AKT];
  __shared__ float sMb[2][AKT];

  const int tid  = threadIdx.x;
  const int lane = tid & 31;
  const int wave = tid >> 5;
  const int hl   = lane >> 4;
  const int l16  = lane & 15;

  const int qt = blockIdx.x;
  const int h  = blockIdx.y;
  const int b  = blockIdx.z;
  const int g  = h & (GROUPS - 1);   // torch repeat-tile semantics: head -> h % G
  const int bg = b * GROUPS + g;

  const int q0 = qt * AQT;
  const size_t qrow0 = (size_t)b * SEQ + q0 + wave * 16;

  // staging geometry
  const int krow = tid >> 1;                 // 0..63 kv row
  const int kseg = (tid & 1) * 64;           // element offset within 128-d row
  const unsigned short* gKbase =
      Kb + ((size_t)b * SEQ + krow) * KV_DIM + g * HEAD_DIM + kseg;
  const unsigned short* gVbase = Vt + ((size_t)bg * HEAD_DIM + tid) * SEQ;
  const unsigned kOff = (unsigned)(krow * HEAD_DIM + kseg) * 2u;
  const unsigned vOff = (unsigned)(tid * AKT) * 2u;

  // Q fragments in registers for the whole loop
  Frag16 qf[4];
  {
    const unsigned short* qp = Qb + (qrow0 + l16) * HIDDEN + h * HEAD_DIM;
#pragma unroll
    for (int dc = 0; dc < 4; ++dc) {
      qf[dc].u4[0] = *(const uint4*)(qp + dc * 32 + hl * 8);
      qf[dc].u4[1] = *(const uint4*)(qp + dc * 32 + 16 + hl * 8);
    }
  }

  v8f o[8] = {};
  float m_i[8], l_i[8];
#pragma unroll
  for (int i = 0; i < 8; ++i) { m_i[i] = -3.0e38f; l_i[i] = 0.0f; }
  const float scale = 0.08838834764831845f;  // 1/sqrt(128)

  // prologue stage kt = 0 into buffer 0
  {
    unsigned lk = lds_off(&sK[0][0]) + kOff;
    unsigned lv = lds_off(&sV[0][0]) + vOff;
#pragma unroll
    for (int i = 0; i < 4; ++i) {
      async_b128(lk + i * 16, gKbase + i * 8);
      async_b128(lv + i * 16, gVbase + i * 8);
    }
    if (tid < AKT) sMb[0][tid] = mask[b * SEQ + tid] ? 0.0f : -1.0e30f;
  }

  int cur = 0;
  for (int kt = 0; kt < SEQ; kt += AKT) {
    wait_async0();
    __syncthreads();

    if (kt + AKT < SEQ) {
      int nxt = kt + AKT;
      unsigned lk = lds_off(&sK[cur ^ 1][0]) + kOff;
      unsigned lv = lds_off(&sV[cur ^ 1][0]) + vOff;
      const unsigned short* gk = gKbase + (size_t)nxt * KV_DIM;
      const unsigned short* gv = gVbase + nxt;
#pragma unroll
      for (int i = 0; i < 4; ++i) {
        async_b128(lk + i * 16, gk + i * 8);
        async_b128(lv + i * 16, gv + i * 8);
      }
      if (tid < AKT) sMb[cur ^ 1][tid] = mask[b * SEQ + nxt + tid] ? 0.0f : -1.0e30f;
    }

    const unsigned short* cK = sK[cur];
    const unsigned short* cV = sV[cur];

    // ---- S = Q @ K^T : dc-outer so the 4 WMMAs per chunk hit independent
    //      accumulators; 4 B-fragments batched ahead of each burst ----
    v8f sc[4] = {};
#pragma unroll
    for (int dc = 0; dc < 4; ++dc) {
      Frag16 kb[4];
#pragma unroll
      for (int nt = 0; nt < 4; ++nt) {
        int n = nt * 16 + l16;
        kb[nt].u4[0] = *(const uint4*)(cK + n * HEAD_DIM + dc * 32 + hl * 8);
        kb[nt].u4[1] = *(const uint4*)(cK + n * HEAD_DIM + dc * 32 + 16 + hl * 8);
      }
#pragma unroll
      for (int nt = 0; nt < 4; ++nt)
        sc[nt] = wmma_bf16(qf[dc], kb[nt], sc[nt]);
    }

    // ---- online softmax update ----
    float pnew[4][8];
#pragma unroll
    for (int i = 0; i < 8; ++i) {
      float s[4];
      float rm = -3.0e38f;
#pragma unroll
      for (int nt = 0; nt < 4; ++nt) {
        s[nt] = sc[nt][i] * scale + sMb[cur][nt * 16 + l16];
        rm = fmaxf(rm, s[nt]);
      }
#pragma unroll
      for (int off = 1; off < 16; off <<= 1)
        rm = fmaxf(rm, __shfl_xor(rm, off, 32));
      float mnew = fmaxf(m_i[i], rm);
      float corr = __expf(m_i[i] - mnew);
      float lsum = 0.0f;
#pragma unroll
      for (int nt = 0; nt < 4; ++nt) {
        float p = __expf(s[nt] - mnew);
        pnew[nt][i] = p;
        lsum += p;
      }
#pragma unroll
      for (int off = 1; off < 16; off <<= 1)
        lsum += __shfl_xor(lsum, off, 32);
      l_i[i] = l_i[i] * corr + lsum;
      m_i[i] = mnew;
#pragma unroll
      for (int nt2 = 0; nt2 < 8; ++nt2) o[nt2][i] *= corr;
    }

    // ---- P (C layout) -> LDS -> A-layout fragments ----
#pragma unroll
    for (int nt = 0; nt < 4; ++nt)
#pragma unroll
      for (int i = 0; i < 8; ++i)
        sP[wave][(hl * 8 + i) * AKT + nt * 16 + l16] = f32_to_bf16(pnew[nt][i]);
    __syncthreads();

    Frag16 pf[2];
#pragma unroll
    for (int kc = 0; kc < 2; ++kc) {
      pf[kc].u4[0] = *(const uint4*)(&sP[wave][l16 * AKT + kc * 32 + hl * 8]);
      pf[kc].u4[1] = *(const uint4*)(&sP[wave][l16 * AKT + kc * 32 + 16 + hl * 8]);
    }

    // ---- O += P @ V : batched fragment loads, bursts of 4 independent WMMAs ----
#pragma unroll
    for (int kc = 0; kc < 2; ++kc)
#pragma unroll
      for (int ng = 0; ng < 2; ++ng) {
        Frag16 vbf[4];
#pragma unroll
        for (int j = 0; j < 4; ++j) {
          int n = (ng * 4 + j) * 16 + l16;
          vbf[j].u4[0] = *(const uint4*)(cV + n * AKT + kc * 32 + hl * 8);
          vbf[j].u4[1] = *(const uint4*)(cV + n * AKT + kc * 32 + 16 + hl * 8);
        }
#pragma unroll
        for (int j = 0; j < 4; ++j)
          o[ng * 4 + j] = wmma_bf16(pf[kc], vbf[j], o[ng * 4 + j]);
      }
    cur ^= 1;
  }

  // ---- epilogue: normalize, store bf16 ----
#pragma unroll
  for (int i = 0; i < 8; ++i) {
    float inv = 1.0f / l_i[i];
    size_t row = qrow0 + hl * 8 + i;
#pragma unroll
    for (int nt = 0; nt < 8; ++nt)
      Ob[row * HIDDEN + h * HEAD_DIM + nt * 16 + l16] = f32_to_bf16(o[nt][i] * inv);
  }
}

// ---------------------------------------------------------------------------
// Host-side launch
// ---------------------------------------------------------------------------
extern "C" void kernel_launch(void* const* d_in, const int* in_sizes, int n_in,
                              void* d_out, int out_size, void* d_ws, size_t ws_size,
                              hipStream_t stream) {
  const float* X  = (const float*)d_in[0];
  const int*   am = (const int*)d_in[1];
  const float* Wq = (const float*)d_in[2];
  const float* bq = (const float*)d_in[3];
  const float* Wk = (const float*)d_in[4];
  const float* bk = (const float*)d_in[5];
  const float* Wv = (const float*)d_in[6];
  const float* bv = (const float*)d_in[7];
  const float* Wo = (const float*)d_in[8];
  const float* bo = (const float*)d_in[9];
  float* out = (float*)d_out;

  char* ws = (char*)d_ws;
  const size_t szX  = (size_t)MROWS * HIDDEN * 2;    // 16 MB
  const size_t szWq = (size_t)HIDDEN * HIDDEN * 2;   //  8 MB
  const size_t szWk = (size_t)HIDDEN * KV_DIM * 2;   //  2 MB
  const size_t szK  = (size_t)MROWS * KV_DIM * 2;    //  4 MB

  size_t off = 0;
  unsigned short* Xb  = (unsigned short*)(ws + off); off += szX;
  unsigned short* Wqt = (unsigned short*)(ws + off); off += szWq;  // [N=2048, K=2048]
  unsigned short* Wkt = (unsigned short*)(ws + off); off += szWk;  // [512, 2048]
  unsigned short* Wvt = (unsigned short*)(ws + off); off += szWk;
  unsigned short* Wot = (unsigned short*)(ws + off); off += szWq;
  unsigned short* Qb  = (unsigned short*)(ws + off); off += szX;
  unsigned short* Kb  = (unsigned short*)(ws + off); off += szK;
  unsigned short* Vb  = (unsigned short*)(ws + off); off += szK;
  unsigned short* Vtb = (unsigned short*)(ws + off); off += szK;
  unsigned short* Ob  = (unsigned short*)(ws + off); off += szX;
  (void)ws_size; (void)in_sizes; (void)n_in; (void)out_size;

  // 1) conversions (weights transposed to [N,K])
  cvt_bf16_kernel<<<2048, 256, 0, stream>>>(X, Xb, (long long)MROWS * HIDDEN / 4);
  cvt_t_bf16_kernel<<<dim3(HIDDEN / 32, HIDDEN / 32), 256, 0, stream>>>(Wq, Wqt, HIDDEN, HIDDEN);
  cvt_t_bf16_kernel<<<dim3(KV_DIM / 32, HIDDEN / 32), 256, 0, stream>>>(Wk, Wkt, HIDDEN, KV_DIM);
  cvt_t_bf16_kernel<<<dim3(KV_DIM / 32, HIDDEN / 32), 256, 0, stream>>>(Wv, Wvt, HIDDEN, KV_DIM);
  cvt_t_bf16_kernel<<<dim3(HIDDEN / 32, HIDDEN / 32), 256, 0, stream>>>(Wo, Wot, HIDDEN, HIDDEN);

  // 2) projections (bf16 out)
  gemm_bf16_kernel<<<dim3(HIDDEN / BN, MROWS / BM), 256, 0, stream>>>(
      Xb, Wqt, bq, Qb, nullptr, MROWS, HIDDEN, HIDDEN);
  gemm_bf16_kernel<<<dim3(KV_DIM / BN, MROWS / BM), 256, 0, stream>>>(
      Xb, Wkt, bk, Kb, nullptr, MROWS, KV_DIM, HIDDEN);
  gemm_bf16_kernel<<<dim3(KV_DIM / BN, MROWS / BM), 256, 0, stream>>>(
      Xb, Wvt, bv, Vb, nullptr, MROWS, KV_DIM, HIDDEN);

  // 3) V -> Vt[b,g,d,S] for async row-contiguous V-tile staging
  vtrans_kernel<<<dim3(SEQ / 32, HEAD_DIM / 32, BATCH * GROUPS), 256, 0, stream>>>(Vb, Vtb);

  // 4) flash attention
  attn_kernel<<<dim3(SEQ / AQT, HEADS, BATCH), 128, 0, stream>>>(Qb, Kb, Vtb, am, Ob);

  // 5) output projection (f32 to d_out)
  gemm_bf16_kernel<<<dim3(HIDDEN / BN, MROWS / BM), 256, 0, stream>>>(
      Ob, Wot, bo, nullptr, out, MROWS, HIDDEN, HIDDEN);
}
